// GraphormerLayer_85237920956532
// MI455X (gfx1250) — matrix operand order repaired and checked by reading, not compiled, not run on hardware
//
#include <hip/hip_runtime.h>
#include <hip/hip_bf16.h>
#include <stdint.h>

// ---------------------------------------------------------------------------
// GraphormerLayer on gfx1250 (MI455X):
//   LN1 -> QKV WMMA GEMMs -> flash attention (+bias, online softmax)
//   -> Wo GEMM + residual -> LN2 -> SiLU FFN (W1,W2) + residual.
// Matrix math: v_wmma_f32_16x16x32_bf16 (wave32 WMMA).
// Staging: global_load_async_to_lds_b128 (ASYNCcnt) + ping-pong LDS; all
// operands stored fragment-major so every frag load is 2x ds_load_b128.
// ---------------------------------------------------------------------------

typedef __attribute__((ext_vector_type(16))) __bf16 v16bf;
typedef __attribute__((ext_vector_type(8)))  float  v8f;

union alignas(32) Frag16 { unsigned int u[8]; uint4 q[2]; v16bf v; };

__device__ __forceinline__ unsigned short f2bfbits(float f) {
  union { float f; unsigned int u; } c; c.f = f;
  unsigned int u = c.u;
  return (unsigned short)((u + 0x7FFFu + ((u >> 16) & 1u)) >> 16); // RNE
}

__device__ __forceinline__ v8f wmma_bf16(const v16bf& a, const v16bf& b, v8f c) {
  return __builtin_amdgcn_wmma_f32_16x16x32_bf16(false, a, false, b, (short)0, c,
                                                 false, false);
}

// low 32 bits of a generic LDS pointer == LDS byte offset (ISA flat->LDS map)
__device__ __forceinline__ unsigned lds_off(const void* p) {
  return (unsigned)(uintptr_t)p;
}

// async DMA: 16B global -> LDS, tracked by ASYNCcnt (no VGPR round trip)
__device__ __forceinline__ void async_b128(unsigned lds, const void* g) {
  asm volatile("global_load_async_to_lds_b128 %0, %1, off"
               :: "v"(lds), "v"((unsigned long long)(uintptr_t)g)
               : "memory");
}
#define WAIT_ASYNCCNT(n) asm volatile("s_wait_asynccnt " #n ::: "memory")

// ------------------------------ LayerNorm ----------------------------------
__global__ __launch_bounds__(256) void ln_norm_k(const float* __restrict__ x,
                                                 const float* __restrict__ g,
                                                 const float* __restrict__ bb,
                                                 unsigned short* __restrict__ out) {
  const int row = blockIdx.x;
  const int tid = threadIdx.x;
  const float4 v = ((const float4*)(x + (size_t)row * 1024))[tid];
  float s  = v.x + v.y + v.z + v.w;
  float s2 = v.x * v.x + v.y * v.y + v.z * v.z + v.w * v.w;
#pragma unroll
  for (int m = 16; m >= 1; m >>= 1) { s += __shfl_xor(s, m); s2 += __shfl_xor(s2, m); }
  __shared__ float red[2][8];
  if ((tid & 31) == 0) { red[0][tid >> 5] = s; red[1][tid >> 5] = s2; }
  __syncthreads();
  float ts = 0.f, ts2 = 0.f;
#pragma unroll
  for (int i = 0; i < 8; ++i) { ts += red[0][i]; ts2 += red[1][i]; }
  const float mu = ts * (1.0f / 1024.0f);
  const float var = ts2 * (1.0f / 1024.0f) - mu * mu;
  const float rs = rsqrtf(var + 1e-5f);
  const int i0 = tid * 4;
  const float xv[4] = { v.x, v.y, v.z, v.w };
  unsigned short* o = out + (size_t)row * 1024 + i0;
#pragma unroll
  for (int k = 0; k < 4; ++k)
    o[k] = f2bfbits((xv[k] - mu) * rs * g[i0 + k] + bb[i0 + k]);
}

// ------------- weight pre-pack: f32 -> fragment-major bf16 pairs ------------
// Wf[((k>>5)*N + c)*16 + ((k&31)>>1)] = {bf16(W[k_even][c]), bf16(W[k_odd][c])}
// Per (k-block, column) the 16 pair-dwords are contiguous (64B) in global.
__global__ __launch_bounds__(256) void pack_w_k(const float* __restrict__ W,
                                                unsigned int* __restrict__ Wf,
                                                int N, int nsh) {
  const size_t idx = (size_t)blockIdx.x * 256 + threadIdx.x; // (kb*16+q)*N + c
  const int c = (int)(idx & (size_t)(N - 1));
  const size_t rest = idx >> nsh;
  const int q = (int)(rest & 15);
  const size_t kb = rest >> 4;
  const size_t k = kb * 32 + q * 2;
  const float lo = W[k * (size_t)N + c];
  const float hi = W[(k + 1) * (size_t)N + c];
  Wf[((kb * N + c) << 4) + q] =
      (unsigned)f2bfbits(lo) | ((unsigned)f2bfbits(hi) << 16);
}

// ------------------------------ WMMA GEMM ----------------------------------
// block tile 64x128, BK=32, 8 waves x (32x32 = 2x2 wmma frags).
#define GM_MODE_HEADQ 0  // bf16 -> (B,H,T,HD)          (used for Q and K)
#define GM_MODE_HEADV 1  // bf16 -> d-major pair-packed Vg[(bh*64+d)*1024+tp]
#define GM_MODE_RESID 2  // f32 out = resid + acc + bias
#define GM_MODE_SILU  3  // bf16 out = silu(acc + bias)

template <int MODE>
__global__ __launch_bounds__(256) void gemm_wmma_k(
    const unsigned short* __restrict__ A, const unsigned int* __restrict__ Wf,
    const float* __restrict__ bias, const float* __restrict__ resid,
    void* __restrict__ outp, int N, int K) {
  __shared__ __align__(16) unsigned int As[2][64 * 20];   // [row][16+4 pad]
  __shared__ __align__(16) unsigned int Bf[2][128 * 20];  // [col][16+4 pad]

  const int tid  = threadIdx.x;
  const int lane = tid & 31;
  const int wave = tid >> 5;
  const int wm = wave >> 2, wn = wave & 3;
  const int l15 = lane & 15, hh = lane >> 4;
  const int bn0 = blockIdx.x * 128;
  const int bm0 = blockIdx.y * 64;

  const v8f zero = {0.f, 0.f, 0.f, 0.f, 0.f, 0.f, 0.f, 0.f};
  v8f acc[2][2];
#pragma unroll
  for (int i = 0; i < 2; ++i)
#pragma unroll
    for (int j = 0; j < 2; ++j) acc[i][j] = zero;

  // async staging coordinates (fixed per thread)
  const int arow = tid >> 2, apc = (tid & 3) * 4;  // A: 1x b128
  const int bcol = tid >> 1, bh8 = (tid & 1) * 8;  // B: 2x b128
  const unsigned int* Apair = (const unsigned int*)A;
  const int Ku = K >> 1;
  const int nt = K >> 5;

  auto issue = [&](int it) {
    const int buf = it & 1;
    async_b128(lds_off(&As[buf][arow * 20 + apc]),
               Apair + (size_t)(bm0 + arow) * Ku + it * 16 + apc);
    const unsigned int* wsrc = Wf + (((size_t)it * N + bn0 + bcol) << 4) + bh8;
    const unsigned bl = lds_off(&Bf[buf][bcol * 20 + bh8]);
    async_b128(bl, wsrc);
    async_b128(bl + 16, wsrc + 4);
  };

  issue(0);
  for (int it = 0; it < nt; ++it) {
    __syncthreads();                       // prev compute done before reuse
    if (it + 1 < nt) { issue(it + 1); WAIT_ASYNCCNT(3); }
    else             { WAIT_ASYNCCNT(0); }
    __syncthreads();                       // tile visible to all waves
    const int buf = it & 1;

    Frag16 af[2], bf[2];
#pragma unroll
    for (int mf = 0; mf < 2; ++mf) {
      const int row = wm * 32 + mf * 16 + l15;
      af[mf].q[0] = *(const uint4*)&As[buf][row * 20 + hh * 8];
      af[mf].q[1] = *(const uint4*)&As[buf][row * 20 + hh * 8 + 4];
    }
#pragma unroll
    for (int nf = 0; nf < 2; ++nf) {
      const int col = wn * 32 + nf * 16 + l15;
      bf[nf].q[0] = *(const uint4*)&Bf[buf][col * 20 + hh * 8];
      bf[nf].q[1] = *(const uint4*)&Bf[buf][col * 20 + hh * 8 + 4];
    }
#pragma unroll
    for (int mf = 0; mf < 2; ++mf)
#pragma unroll
      for (int nf = 0; nf < 2; ++nf)
        acc[mf][nf] = wmma_bf16(af[mf].v, bf[nf].v, acc[mf][nf]);
  }

  // --- epilogue (compile-time specialized) ---
  unsigned short* outh = (unsigned short*)outp;
  float* outf = (float*)outp;
#pragma unroll
  for (int mf = 0; mf < 2; ++mf)
#pragma unroll
    for (int nf = 0; nf < 2; ++nf)
#pragma unroll
      for (int e = 0; e < 8; ++e) {
        const int r = bm0 + wm * 32 + mf * 16 + hh * 8 + e;
        const int c = bn0 + wn * 32 + nf * 16 + l15;
        const float val = acc[mf][nf][e] + bias[c];
        if (MODE == GM_MODE_HEADQ) {
          const int bi = r >> 11, t = r & 2047, hd = c >> 6, d = c & 63;
          outh[(((size_t)(bi * 16 + hd)) * 2048 + t) * 64 + d] = f2bfbits(val);
        } else if (MODE == GM_MODE_HEADV) {
          const int bi = r >> 11, t = r & 2047, hd = c >> 6, d = c & 63;
          const int bh = bi * 16 + hd;
          outh[(((size_t)(bh * 64 + d)) * 1024 + (t >> 1)) * 2 + (t & 1)] = f2bfbits(val);
        } else if (MODE == GM_MODE_RESID) {
          const size_t idx = (size_t)r * N + c;
          outf[idx] = resid[idx] + val;
        } else {
          const float sg = val / (1.0f + __expf(-val));
          outh[(size_t)r * N + c] = f2bfbits(sg);
        }
      }
}

// --------------------------- Flash attention -------------------------------
// grid=(B*H, T/128); 8 waves x 16 query rows. K (token-major, plain QKV
// layout) and V (d-major pair-packed) tiles async-DMAed contiguously into
// LDS, double buffered; every fragment load is 2x ds_load_b128.
__global__ __launch_bounds__(256) void flash_attn_k(
    const unsigned short* __restrict__ qbuf, const unsigned short* __restrict__ kbuf,
    const unsigned int* __restrict__ vg, const float* __restrict__ bias,
    unsigned short* __restrict__ obuf) {
  __shared__ __align__(16) unsigned int Kt[2][64 * 40];   // [token][32+8 pad]
  __shared__ __align__(16) unsigned int Vt[2][64 * 40];   // [d][32+8 pad]
  __shared__ __align__(16) unsigned int Pp[8 * 16 * 36];  // per-wave P tiles

  const int tid  = threadIdx.x;
  const int lane = tid & 31;
  const int wave = tid >> 5;
  const int l15 = lane & 15, hh = lane >> 4;
  const int bh = blockIdx.x;            // b*16 + h
  const int b  = bh >> 4, h = bh & 15;
  const int trow0 = blockIdx.y * 128 + wave * 16;
  const size_t qBase = (size_t)bh * 2048 * 64;
  const size_t biasBase = (size_t)bh * 2048 * 2048;
  const float scale = 0.125f; // 1/sqrt(64)

  // Q fragments: 16 rows x 64 d, in registers (2x global b128 per frag)
  Frag16 qf[2];
  {
    const unsigned int* qrow =
        (const unsigned int*)(qbuf + qBase + (size_t)(trow0 + l15) * 64);
#pragma unroll
    for (int kf = 0; kf < 2; ++kf) {
      qf[kf].q[0] = *(const uint4*)(qrow + kf * 16 + hh * 8);
      qf[kf].q[1] = *(const uint4*)(qrow + kf * 16 + hh * 8 + 4);
    }
  }

  const v8f zero = {0.f, 0.f, 0.f, 0.f, 0.f, 0.f, 0.f, 0.f};
  v8f oacc[4];
#pragma unroll
  for (int df = 0; df < 4; ++df) oacc[df] = zero;
  float rmax[8], rsum[8];
#pragma unroll
  for (int e = 0; e < 8; ++e) { rmax[e] = -1e30f; rsum[e] = 0.f; }

  unsigned short* PpH = (unsigned short*)(Pp + wave * 16 * 36);
  const unsigned int* kU = (const unsigned int*)kbuf;

  // async K/V tile staging: 4x b128 per thread per 64-token tile
  const int sr = tid >> 2, sp0 = (tid & 3) * 8;
  auto issue = [&](int it) {
    const int buf = it & 1;
    const int j = it * 64;
    const unsigned int* ks = kU + ((size_t)bh * 2048 + j + sr) * 32 + sp0;
    const unsigned kl = lds_off(&Kt[buf][sr * 40 + sp0]);
    async_b128(kl, ks);
    async_b128(kl + 16, ks + 4);
    const unsigned int* vs = vg + ((size_t)(bh * 64 + sr)) * 1024 + (j >> 1) + sp0;
    const unsigned vl = lds_off(&Vt[buf][sr * 40 + sp0]);
    async_b128(vl, vs);
    async_b128(vl + 16, vs + 4);
  };

  issue(0);
  for (int it = 0; it < 32; ++it) {
    __syncthreads();
    if (it + 1 < 32) { issue(it + 1); WAIT_ASYNCCNT(4); }
    else             { WAIT_ASYNCCNT(0); }
    __syncthreads();
    const int buf = it & 1;
    const int j = it * 64;

    // --- S = Q K^T * scale + bias ---
    float sv[4][8];
#pragma unroll
    for (int cf = 0; cf < 4; ++cf) {
      const int tcol = cf * 16 + l15;
      Frag16 bf0, bf1;
      bf0.q[0] = *(const uint4*)&Kt[buf][tcol * 40 + hh * 8];
      bf0.q[1] = *(const uint4*)&Kt[buf][tcol * 40 + hh * 8 + 4];
      bf1.q[0] = *(const uint4*)&Kt[buf][tcol * 40 + 16 + hh * 8];
      bf1.q[1] = *(const uint4*)&Kt[buf][tcol * 40 + 16 + hh * 8 + 4];
      v8f s = zero;
      s = wmma_bf16(qf[0].v, bf0.v, s);
      s = wmma_bf16(qf[1].v, bf1.v, s);
      const int cg = j + cf * 16 + l15;
#pragma unroll
      for (int e = 0; e < 8; ++e) {
        const int tr = trow0 + hh * 8 + e;
        sv[cf][e] = s[e] * scale + bias[biasBase + (size_t)tr * 2048 + cg];
      }
    }

    // --- online softmax (row reductions via shfl_xor within 16 lanes) ---
#pragma unroll
    for (int e = 0; e < 8; ++e) {
      float m = fmaxf(fmaxf(sv[0][e], sv[1][e]), fmaxf(sv[2][e], sv[3][e]));
#pragma unroll
      for (int msk = 8; msk >= 1; msk >>= 1) m = fmaxf(m, __shfl_xor(m, msk));
      const float nm = fmaxf(rmax[e], m);
      const float corr = __expf(rmax[e] - nm);
      rmax[e] = nm;
      float ls = 0.f;
#pragma unroll
      for (int cf = 0; cf < 4; ++cf) {
        const float p = __expf(sv[cf][e] - nm);
        sv[cf][e] = p;
        ls += p;
      }
#pragma unroll
      for (int msk = 8; msk >= 1; msk >>= 1) ls += __shfl_xor(ls, msk);
      rsum[e] = rsum[e] * corr + ls;
#pragma unroll
      for (int df = 0; df < 4; ++df) oacc[df][e] *= corr;
    }

    // --- transpose P (C-layout -> A-frag layout) via per-wave LDS tile ---
#pragma unroll
    for (int cf = 0; cf < 4; ++cf)
#pragma unroll
      for (int e = 0; e < 8; ++e)
        PpH[(hh * 8 + e) * 72 + cf * 16 + l15] = f2bfbits(sv[cf][e]);

    Frag16 pf[2];
#pragma unroll
    for (int kf = 0; kf < 2; ++kf) {
      const unsigned int* pr = &Pp[wave * 16 * 36 + l15 * 36 + kf * 16];
      pf[kf].q[0] = *(const uint4*)(pr + hh * 8);
      pf[kf].q[1] = *(const uint4*)(pr + hh * 8 + 4);
    }

    // --- O += P V ---
#pragma unroll
    for (int df = 0; df < 4; ++df) {
      const int dcol = df * 16 + l15;
      Frag16 vf0, vf1;
      vf0.q[0] = *(const uint4*)&Vt[buf][dcol * 40 + hh * 8];
      vf0.q[1] = *(const uint4*)&Vt[buf][dcol * 40 + hh * 8 + 4];
      vf1.q[0] = *(const uint4*)&Vt[buf][dcol * 40 + 16 + hh * 8];
      vf1.q[1] = *(const uint4*)&Vt[buf][dcol * 40 + 16 + hh * 8 + 4];
      oacc[df] = wmma_bf16(pf[0].v, vf0.v, oacc[df]);
      oacc[df] = wmma_bf16(pf[1].v, vf1.v, oacc[df]);
    }
  }

  // --- normalize + write O to (B,T,D) bf16 ---
#pragma unroll
  for (int df = 0; df < 4; ++df)
#pragma unroll
    for (int e = 0; e < 8; ++e) {
      const int t = trow0 + hh * 8 + e;
      const int d = df * 16 + l15;
      const float val = oacc[df][e] / rsum[e];
      obuf[((size_t)(b * 2048 + t)) * 1024 + h * 64 + d] = f2bfbits(val);
    }
}

// ------------------------------ launcher -----------------------------------
extern "C" void kernel_launch(void* const* d_in, const int* in_sizes, int n_in,
                              void* d_out, int out_size, void* d_ws, size_t ws_size,
                              hipStream_t stream) {
  (void)in_sizes; (void)n_in; (void)out_size; (void)ws_size;
  const float* x         = (const float*)d_in[0];
  const float* attn_bias = (const float*)d_in[1];
  const float* ln1_g = (const float*)d_in[2];
  const float* ln1_b = (const float*)d_in[3];
  const float* Wq = (const float*)d_in[4];  const float* bq = (const float*)d_in[5];
  const float* Wk = (const float*)d_in[6];  const float* bk = (const float*)d_in[7];
  const float* Wv = (const float*)d_in[8];  const float* bv = (const float*)d_in[9];
  const float* Wo = (const float*)d_in[10]; const float* bo = (const float*)d_in[11];
  const float* ln2_g = (const float*)d_in[12];
  const float* ln2_b = (const float*)d_in[13];
  const float* W1 = (const float*)d_in[14]; const float* b1 = (const float*)d_in[15];
  const float* W2 = (const float*)d_in[16]; const float* b2 = (const float*)d_in[17];

  char* w = (char*)d_ws;
  const size_t MB = 1 << 20;
  unsigned short* h1 = (unsigned short*)(w + 0 * MB);    //  8 MB
  unsigned short* qb = (unsigned short*)(w + 8 * MB);    //  8 MB (B,H,T,HD)
  unsigned short* kb2 = (unsigned short*)(w + 16 * MB);  //  8 MB (B,H,T,HD)
  unsigned int*   vgp = (unsigned int*)(w + 24 * MB);    //  8 MB d-major pairs
  unsigned short* ob = (unsigned short*)(w + 32 * MB);   //  8 MB
  float*          x2 = (float*)         (w + 40 * MB);   // 16 MB
  unsigned short* h2 = (unsigned short*)(w + 56 * MB);   //  8 MB
  unsigned short* f1 = (unsigned short*)(w + 64 * MB);   // 32 MB
  unsigned int* Wq_p = (unsigned int*)(w + 96 * MB);     //  2 MB
  unsigned int* Wk_p = (unsigned int*)(w + 98 * MB);     //  2 MB
  unsigned int* Wv_p = (unsigned int*)(w + 100 * MB);    //  2 MB
  unsigned int* Wo_p = (unsigned int*)(w + 102 * MB);    //  2 MB
  unsigned int* W1_p = (unsigned int*)(w + 104 * MB);    //  8 MB
  unsigned int* W2_p = (unsigned int*)(w + 112 * MB);    //  8 MB

  const dim3 blk(256);
  const dim3 gD(1024 / 128, 4096 / 64);   // N=1024 GEMMs
  const dim3 gF(4096 / 128, 4096 / 64);   // N=4096 GEMM (W1)

  // 0) pre-pack weights to fragment-major bf16 pairs
  pack_w_k<<<2048, blk, 0, stream>>>(Wq, Wq_p, 1024, 10);
  pack_w_k<<<2048, blk, 0, stream>>>(Wk, Wk_p, 1024, 10);
  pack_w_k<<<2048, blk, 0, stream>>>(Wv, Wv_p, 1024, 10);
  pack_w_k<<<2048, blk, 0, stream>>>(Wo, Wo_p, 1024, 10);
  pack_w_k<<<8192, blk, 0, stream>>>(W1, W1_p, 4096, 12);
  pack_w_k<<<8192, blk, 0, stream>>>(W2, W2_p, 1024, 10);

  // 1) LN1
  ln_norm_k<<<4096, blk, 0, stream>>>(x, ln1_g, ln1_b, h1);
  // 2) QKV projections (V in attention-DMA-friendly d-major layout)
  gemm_wmma_k<GM_MODE_HEADQ><<<gD, blk, 0, stream>>>(h1, Wq_p, bq, nullptr, qb, 1024, 1024);
  gemm_wmma_k<GM_MODE_HEADQ><<<gD, blk, 0, stream>>>(h1, Wk_p, bk, nullptr, kb2, 1024, 1024);
  gemm_wmma_k<GM_MODE_HEADV><<<gD, blk, 0, stream>>>(h1, Wv_p, bv, nullptr, vgp, 1024, 1024);
  // 3) flash attention with additive bias
  flash_attn_k<<<dim3(32, 16), blk, 0, stream>>>(qb, kb2, vgp, attn_bias, ob);
  // 4) output projection + residual -> x2 (f32)
  gemm_wmma_k<GM_MODE_RESID><<<gD, blk, 0, stream>>>(ob, Wo_p, bo, x, x2, 1024, 1024);
  // 5) LN2
  ln_norm_k<<<4096, blk, 0, stream>>>(x2, ln2_g, ln2_b, h2);
  // 6) FFN: silu(h2 @ W1 + b1) -> f1 (bf16)
  gemm_wmma_k<GM_MODE_SILU><<<gF, blk, 0, stream>>>(h2, W1_p, b1, nullptr, f1, 4096, 1024);
  // 7) f1 @ W2 + b2 + x2 -> d_out (f32)
  gemm_wmma_k<GM_MODE_RESID><<<gD, blk, 0, stream>>>(f1, W2_p, b2, x2, (float*)d_out, 1024, 4096);
}
